// TraditionalGNN_35639638622631
// MI455X (gfx1250) — matrix operand order repaired and checked by reading, not compiled, not run on hardware
//
#include <hip/hip_runtime.h>
#include <stddef.h>

// ---------------------------------------------------------------------------
// CDNA5 (gfx1250) GNN forward: mean-agg (atomic scatter) + bf16 WMMA GEMMs.
// N_NODES=100000, N_EDGES=1200000, IN_DIM=6, HID=64.
// ---------------------------------------------------------------------------

typedef __attribute__((ext_vector_type(16))) __bf16 v16bf;
typedef __attribute__((ext_vector_type(8)))  __bf16 v8bf;
typedef __attribute__((ext_vector_type(8)))  float  v8f;
typedef __attribute__((ext_vector_type(4)))  float  v4f;

// ---------------------------------------------------------------- fill zeros
__global__ __launch_bounds__(256) void gnn_fill0_f32(float* __restrict__ p, long long cnt) {
    long long t = (long long)blockIdx.x * blockDim.x + threadIdx.x;
    if (t < cnt) p[t] = 0.0f;
}

// ---------------------------------------------------------------- degree
__global__ __launch_bounds__(256) void gnn_degree(const long long* __restrict__ ei,
                                                  float* __restrict__ deg, int nEdges) {
    int e = blockIdx.x * blockDim.x + threadIdx.x;
    if (e >= nEdges) return;
    int dst = (int)ei[e];                       // row = edge_index[0]
    atomicAdd(&deg[dst], 1.0f);
}

__global__ __launch_bounds__(256) void gnn_deginv(float* __restrict__ deg, int n) {
    int i = blockIdx.x * blockDim.x + threadIdx.x;
    if (i >= n) return;
    deg[i] = 1.0f / fmaxf(deg[i], 1.0f);        // in-place: deg -> deg_inv
}

// ------------------------------------------------- scatter-sum, 6-wide input
__global__ __launch_bounds__(256) void gnn_scatter6(const long long* __restrict__ ei,
                                                    const float* __restrict__ x,
                                                    float* __restrict__ agg, int nEdges) {
    int e = blockIdx.x * blockDim.x + threadIdx.x;
    if (e >= nEdges) return;
    int dst = (int)ei[e];
    int src = (int)ei[nEdges + e];
    const float* s = x + (size_t)src * 6;
    float* o = agg + (size_t)dst * 6;
#pragma unroll
    for (int d = 0; d < 6; ++d) atomicAdd(&o[d], s[d]);
}

// scale by deg_inv, convert to bf16, zero-pad 6 -> 64 columns
__global__ __launch_bounds__(256) void gnn_cvt6(const float* __restrict__ agg6,
                                                const float* __restrict__ deg_inv,
                                                __bf16* __restrict__ aggb, int n) {
    long long t = (long long)blockIdx.x * blockDim.x + threadIdx.x;
    if (t >= (long long)n * 64) return;
    int i = (int)(t >> 6), d = (int)(t & 63);
    float v = (d < 6) ? agg6[(size_t)i * 6 + d] * deg_inv[i] : 0.0f;
    aggb[t] = (__bf16)v;
}

// ------------------------------------------------- scatter-sum, 64-wide feat
// 16 threads per edge, each does a float4 gather + 4 L2 atomics.
__global__ __launch_bounds__(256) void gnn_scatter64(const long long* __restrict__ ei,
                                                     const float* __restrict__ h,
                                                     float* __restrict__ agg, int nEdges) {
    long long t = (long long)blockIdx.x * blockDim.x + threadIdx.x;
    if (t >= (long long)nEdges * 16) return;
    int e  = (int)(t >> 4);
    int d4 = (int)(t & 15) * 4;
    int dst = (int)ei[e];
    int src = (int)ei[nEdges + e];
    const float4 v = *(const float4*)(h + (size_t)src * 64 + d4);
    float* o = agg + (size_t)dst * 64 + d4;
    atomicAdd(o + 0, v.x); atomicAdd(o + 1, v.y);
    atomicAdd(o + 2, v.z); atomicAdd(o + 3, v.w);
}

// scale by deg_inv and convert fp32 -> bf16 (WMMA A staging), 8 elems/thread
__global__ __launch_bounds__(256) void gnn_cvt64(const float* __restrict__ agg,
                                                 const float* __restrict__ deg_inv,
                                                 __bf16* __restrict__ aggb, int n) {
    long long t = (long long)blockIdx.x * blockDim.x + threadIdx.x;  // group of 8
    if (t >= (long long)n * 8) return;
    const float di = deg_inv[t >> 3];          // 8 groups per 64-wide row
    const v4f a0 = *(const v4f*)(agg + t * 8);
    const v4f a1 = *(const v4f*)(agg + t * 8 + 4);
    v8bf o;
#pragma unroll
    for (int i = 0; i < 4; ++i) {
        o[i]     = (__bf16)(a0[i] * di);
        o[4 + i] = (__bf16)(a1[i] * di);
    }
    *(v8bf*)(aggb + t * 8) = o;
}

// W [fanIn x 64] row-major fp32 -> Wt [64(N) x 64(K)] bf16, zero-padded K
__global__ __launch_bounds__(256) void gnn_wcvt(const float* __restrict__ W,
                                                __bf16* __restrict__ Wt, int fanIn) {
    int t = blockIdx.x * blockDim.x + threadIdx.x;
    if (t >= 64 * 64) return;
    int nIdx = t >> 6, k = t & 63;
    float v = (k < fanIn) ? W[k * 64 + nIdx] : 0.0f;
    Wt[t] = (__bf16)v;
}

// ---------------------------------------------------------------- WMMA GEMM
// out[n,64] = relu(A[n,64]_bf16 * W[64,64] + bias) (+ res). One wave owns a
// 16x16 f32 tile; K=64 consumed by two v_wmma_f32_16x16x32_bf16.
// Block = 8 waves = 2 row-tiles x 4 col-tiles. HAS_RES is compile-time so the
// epilogue fast path is branch-free (8 immediate-offset b32 stores).
template <bool HAS_RES>
__global__ __launch_bounds__(256) void gnn_gemm64_wmma(const __bf16* __restrict__ A,
                                                       const __bf16* __restrict__ Wt,
                                                       const float* __restrict__ bias,
                                                       const float* __restrict__ res,
                                                       float* __restrict__ out, int n) {
    const int lane    = threadIdx.x & 31;
    const int wave    = threadIdx.x >> 5;
    const int rowBase = (blockIdx.x * 2 + (wave >> 2)) * 16;
    const int colBase = (wave & 3) * 16;
    if (rowBase >= n) return;                 // wave-uniform: EXEC stays all-1s

    const int m    = lane & 15;               // A: row-in-tile / B,C,D: N index
    const int hi   = lane >> 4;               // lane half selector
    const int kOff = hi * 8;                  // ISA A-layout: lanes 16-31 hold K+8

    int row = rowBase + m;
    if (row >= n) row = n - 1;                // keep all lanes active on partial tile

    // ---- A fragments (16x32 bf16 each): per-lane K runs of 8 are contiguous
    v16bf a0, a1;
    {
        const __bf16* ar = A + (size_t)row * 64;
        v8bf t0 = *(const v8bf*)(ar + 0  + kOff);   // K = kOff .. kOff+7
        v8bf t1 = *(const v8bf*)(ar + 16 + kOff);   // K = 16+kOff ..
        v8bf t2 = *(const v8bf*)(ar + 32 + kOff);   // chunk 1
        v8bf t3 = *(const v8bf*)(ar + 48 + kOff);
#pragma unroll
        for (int i = 0; i < 8; ++i) {
            a0[i] = t0[i]; a0[8 + i] = t1[i];
            a1[i] = t2[i]; a1[8 + i] = t3[i];
        }
    }
    // ---- B fragments (32x16 bf16): lane holds N=m, K runs of 8 matching A layout (Wt is [N][K])
    v16bf b0, b1;
    {
        const __bf16* br = Wt + (size_t)(colBase + m) * 64 + hi * 8;
        v8bf t0 = *(const v8bf*)(br + 0);
        v8bf t1 = *(const v8bf*)(br + 16);
        v8bf t2 = *(const v8bf*)(br + 32);
        v8bf t3 = *(const v8bf*)(br + 48);
#pragma unroll
        for (int i = 0; i < 8; ++i) {
            b0[i] = t0[i]; b0[8 + i] = t1[i];
            b1[i] = t2[i]; b1[8 + i] = t3[i];
        }
    }

    v8f c = {0.f, 0.f, 0.f, 0.f, 0.f, 0.f, 0.f, 0.f};
    c = __builtin_amdgcn_wmma_f32_16x16x32_bf16(false, a0, false, b0, (short)0, c, false, false);
    c = __builtin_amdgcn_wmma_f32_16x16x32_bf16(false, a1, false, b1, (short)0, c, false, false);

    // ---- epilogue: bias + relu (+ residual). C/D layout: VGPR r -> M=r+8*hi, N=m.
    const int    nCol  = colBase + m;
    const float  bv    = bias[nCol];
    float*       ob    = out + (size_t)(rowBase + hi * 8) * 64 + nCol;
    const float* rb    = HAS_RES ? res + (size_t)(rowBase + hi * 8) * 64 + nCol : nullptr;

    if (rowBase + 16 <= n) {                  // full tile: branch-free fast path
#pragma unroll
        for (int r = 0; r < 8; ++r) {
            float v = fmaxf(c[r] + bv, 0.0f);
            if (HAS_RES) v += rb[(size_t)r * 64];
            ob[(size_t)r * 64] = v;
        }
    } else {                                  // partial last tile (n % 16 != 0)
#pragma unroll
        for (int r = 0; r < 8; ++r) {
            int orow = rowBase + r + hi * 8;
            if (orow < n) {
                float v = fmaxf(c[r] + bv, 0.0f);
                if (HAS_RES) v += rb[(size_t)r * 64];
                ob[(size_t)r * 64] = v;
            }
        }
    }
}

// -------------------------------------------------------- final 64->1 layer
__global__ __launch_bounds__(256) void gnn_final(const float* __restrict__ agg,
                                                 const float* __restrict__ deg_inv,
                                                 const float* __restrict__ W5,
                                                 const float* __restrict__ b5,
                                                 float* __restrict__ out, int n) {
    int i = blockIdx.x * blockDim.x + threadIdx.x;
    if (i >= n) return;
    const float* a = agg + (size_t)i * 64;
    float s = 0.0f;
#pragma unroll 8
    for (int d = 0; d < 64; ++d) s += a[d] * W5[d];
    out[i] = s * deg_inv[i] + b5[0];
}

// ---------------------------------------------------------------------------
extern "C" void kernel_launch(void* const* d_in, const int* in_sizes, int n_in,
                              void* d_out, int out_size, void* d_ws, size_t ws_size,
                              hipStream_t stream) {
    const float*     x  = (const float*)d_in[0];
    const long long* ei = (const long long*)d_in[1];   // int64 edge_index (2, E)
    const float* W[6] = {(const float*)d_in[2],  (const float*)d_in[4],
                         (const float*)d_in[6],  (const float*)d_in[8],
                         (const float*)d_in[10], (const float*)d_in[12]};
    const float* B[6] = {(const float*)d_in[3],  (const float*)d_in[5],
                         (const float*)d_in[7],  (const float*)d_in[9],
                         (const float*)d_in[11], (const float*)d_in[13]};

    const int n      = in_sizes[0] / 6;    // IN_DIM = 6
    const int nEdges = in_sizes[1] / 2;    // (2, E)

    // ---- carve workspace (256B aligned)
    char* ws = (char*)d_ws;
    auto take = [&](size_t bytes) {
        char* p = ws;
        ws += (bytes + 255) & ~(size_t)255;
        return p;
    };
    float*  deg_inv = (float*)take((size_t)n * 4);            // deg -> deg_inv
    float*  agg     = (float*)take((size_t)n * 64 * 4);       // fp32 scatter target
    __bf16* aggb    = (__bf16*)take((size_t)n * 64 * 2);      // WMMA A staging
    float*  hA      = (float*)take((size_t)n * 64 * 4);       // ping
    float*  hB      = (float*)take((size_t)n * 64 * 4);       // pong
    __bf16* Wt      = (__bf16*)take((size_t)5 * 64 * 64 * 2); // bf16 weights
    (void)ws_size;

    const int T = 256;
    auto blk  = [&](long long c) { return (unsigned)((c + T - 1) / T); };
    const unsigned gN    = blk(n);
    const unsigned gN64  = blk((long long)n * 64);
    const unsigned gN8   = blk((long long)n * 8);
    const unsigned gE    = blk(nEdges);
    const unsigned gE16  = blk((long long)nEdges * 16);
    const unsigned gGemm = (unsigned)((n + 31) / 32);         // 2 row-tiles / block

    // ---- weights to bf16 [N][K] (zero-padded K)
    for (int l = 0; l < 5; ++l)
        gnn_wcvt<<<16, T, 0, stream>>>(W[l], Wt + (size_t)l * 4096, l == 0 ? 6 : 64);

    // ---- degree / deg_inv
    gnn_fill0_f32<<<gN, T, 0, stream>>>(deg_inv, n);
    gnn_degree<<<gE, T, 0, stream>>>(ei, deg_inv, nEdges);
    gnn_deginv<<<gN, T, 0, stream>>>(deg_inv, n);

    // ---- layer 0: agg(x) [n,6] -> bf16 [n,64] padded -> WMMA -> hA
    gnn_fill0_f32<<<blk((long long)n * 6), T, 0, stream>>>(agg, (long long)n * 6);
    gnn_scatter6<<<gE, T, 0, stream>>>(ei, x, agg, nEdges);
    gnn_cvt6<<<gN64, T, 0, stream>>>(agg, deg_inv, aggb, n);
    gnn_gemm64_wmma<false><<<gGemm, T, 0, stream>>>(aggb, Wt, B[0], nullptr, hA, n);

    // ---- layers 1..4: h = relu(mean_agg(h) @ Wl + bl) + h
    float* hCur = hA;
    float* hNxt = hB;
    for (int l = 1; l <= 4; ++l) {
        gnn_fill0_f32<<<gN64, T, 0, stream>>>(agg, (long long)n * 64);
        gnn_scatter64<<<gE16, T, 0, stream>>>(ei, hCur, agg, nEdges);
        gnn_cvt64<<<gN8, T, 0, stream>>>(agg, deg_inv, aggb, n);
        gnn_gemm64_wmma<true><<<gGemm, T, 0, stream>>>(aggb, Wt + (size_t)l * 4096, B[l], hCur, hNxt, n);
        float* t = hCur; hCur = hNxt; hNxt = t;
    }

    // ---- final: mean_agg(h) @ W5 + b5  (64->1, fp32 scalar)
    gnn_fill0_f32<<<gN64, T, 0, stream>>>(agg, (long long)n * 64);
    gnn_scatter64<<<gE16, T, 0, stream>>>(ei, hCur, agg, nEdges);
    gnn_final<<<gN, T, 0, stream>>>(agg, deg_inv, W[5], B[5], (float*)d_out, n);
}
